// FermiDiracDecoder_21655225107257
// MI455X (gfx1250) — matrix operand order repaired and checked by reading, not compiled, not run on hardware
//
#include <hip/hip_runtime.h>

#define NPTS 8192
#define DIM  256
#define BT   128   // block output tile (BT x BT)
#define KS   32    // K step per stage
#define LDT  40    // padded LDS row stride in bf16 elements (80B = 64B data + 16B pad)
#define NSTEP (DIM / KS)

typedef __attribute__((ext_vector_type(16))) __bf16 v16bf;
typedef __attribute__((ext_vector_type(8)))  float  v8f;
typedef __attribute__((ext_vector_type(4)))  unsigned int v4u;
typedef __attribute__((ext_vector_type(8)))  int v8i;
typedef __attribute__((ext_vector_type(4)))  int v4i;

union Frag {
  v16bf v;
  uint4 q[2];
};

// ---------------- phase 0: row norms + bf16 hi/lo split ----------------
__global__ __launch_bounds__(256) void prep_kernel(
    const float* __restrict__ x,
    unsigned short* __restrict__ hi_u,
    unsigned short* __restrict__ lo_u,
    float* __restrict__ sq)
{
  __bf16* hi = reinterpret_cast<__bf16*>(hi_u);
  __bf16* lo = reinterpret_cast<__bf16*>(lo_u);
  const int lane = threadIdx.x & 31;
  const int wid  = threadIdx.x >> 5;
  const int row  = blockIdx.x * 8 + wid;
  const float* xr = x + (size_t)row * DIM;
  const size_t base = (size_t)row * DIM + lane * 8;
  float s = 0.f;
#pragma unroll
  for (int u = 0; u < 8; ++u) {
    float v = xr[lane * 8 + u];
    s += v * v;
    __bf16 h = (__bf16)v;            // RNE round to bf16
    float  hf = (float)h;
    __bf16 l = (__bf16)(v - hf);     // residual in bf16
    hi[base + u] = h;
    lo[base + u] = l;
  }
#pragma unroll
  for (int m = 16; m >= 1; m >>= 1) s += __shfl_xor(s, m, 32);
  if (lane == 0) sq[row] = s;
}

// ---------------- TDM panel fetch: 128 rows x 64B tile -> padded LDS ----------------
// D# per CDNA5 ISA:
//   data_size=2 (4B units), tile_dim0=16 (64B row), tile_dim1=128,
//   tensor_dim0=16, tensor_dim1=128, tensor_dim0_stride=128 (512B row pitch),
//   pad_enable=1, pad_interval=3 (pad after 16 DWORDs), pad_amount=3 (4 DWORDs = 16B)
__device__ __forceinline__ void tdm_issue(const unsigned short* base, int row0, int k0,
                                          unsigned lds_off)
{
  unsigned long long gaddr = (unsigned long long)(uintptr_t)base
                           + ((unsigned long long)(unsigned)row0 * DIM + (unsigned)k0) * 2ull;
  v4u g0;
  g0.x = 1u;                                             // count=1 (valid), user mode
  g0.y = lds_off;                                        // lds_addr
  g0.z = (unsigned)(gaddr & 0xFFFFFFFFull);              // global_addr[31:0]
  g0.w = (unsigned)((gaddr >> 32) & 0x01FFFFFFull)       // global_addr[56:32]
       | (2u << 30);                                     // type=2 ("image")
  v8i g1;
  g1[0] = (int)((2u << 16)       // data_size = 4B units
              | (1u << 20)       // pad_enable
              | (3u << 22)       // pad_interval: 16 DWORDs (one 64B row)
              | (3u << 25));     // pad_amount: 4 DWORDs (16B)
  g1[1] = (int)(16u << 16);      // tensor_dim0 = 16 (low 16 bits @ [63:48])
  g1[2] = (int)(128u << 16);     // tensor_dim1 = 128 (low 16 bits @ [111:96->95:80])
  g1[3] = (int)(16u << 16);      // tile_dim0 = 16 @ [127:112]
  g1[4] = (int)128;              // tile_dim1 = 128 @ [143:128]
  g1[5] = (int)128;              // tensor_dim0_stride = 128 @ [207:160] low 32
  g1[6] = 0;
  g1[7] = 0;
  v4i z4 = {0, 0, 0, 0};
#if defined(__clang_major__) && (__clang_major__ >= 23)
  v8i z8 = {0, 0, 0, 0, 0, 0, 0, 0};
  __builtin_amdgcn_tensor_load_to_lds(g0, g1, z4, z4, z8, 0);
#else
  __builtin_amdgcn_tensor_load_to_lds(g0, g1, z4, z4, 0);
#endif
}

// ---------------- phase 1: WMMA GEMM + hyperbolic epilogue ----------------
__global__ __launch_bounds__(256) void fermi_kernel(
    const unsigned short* __restrict__ hi_u,
    const unsigned short* __restrict__ lo_u,
    const float* __restrict__ sq,
    float* __restrict__ out)
{
  // [buf][panel: 0=Ahi 1=Alo 2=Bhi 3=Blo][128 rows * LDT]
  __shared__ __bf16 sPan[2][4][BT * LDT];
  __shared__ float  sSqR[BT];
  __shared__ float  sSqC[BT];

  const int tid  = threadIdx.x;
  const int lane = tid & 31;
  const int wid  = tid >> 5;       // 8 waves
  const int lm   = lane & 15;
  const int half = lane >> 4;
  const int waveM = wid & 3;       // 4 row groups of 32
  const int waveN = wid >> 2;      // 2 col groups of 64
  const int rowBase = blockIdx.y * BT;
  const int colBase = blockIdx.x * BT;

  if (tid < 128) sSqR[tid] = sq[rowBase + tid];
  else           sSqC[tid - 128] = sq[colBase + (tid - 128)];

  v8f acc[2][4];
  const v8f vzero = {0.f, 0.f, 0.f, 0.f, 0.f, 0.f, 0.f, 0.f};
#pragma unroll
  for (int a = 0; a < 2; ++a)
#pragma unroll
    for (int b = 0; b < 4; ++b) acc[a][b] = vzero;

  // prologue: wave 0 DMAs the first K panels, waits, then releases the block
  if (wid == 0) {
    tdm_issue(hi_u, rowBase, 0, (unsigned)(uintptr_t)&sPan[0][0][0]);
    tdm_issue(lo_u, rowBase, 0, (unsigned)(uintptr_t)&sPan[0][1][0]);
    tdm_issue(hi_u, colBase, 0, (unsigned)(uintptr_t)&sPan[0][2][0]);
    tdm_issue(lo_u, colBase, 0, (unsigned)(uintptr_t)&sPan[0][3][0]);
    __builtin_amdgcn_s_wait_tensorcnt(0);
  }
  __syncthreads();

  for (int ks = 0; ks < NSTEP; ++ks) {
    const int cur = ks & 1;
    // prefetch next K panels into the other buffer (overlaps with WMMA below)
    if ((ks + 1) < NSTEP && wid == 0) {
      const int nxt = cur ^ 1;
      const int k0n = (ks + 1) * KS;
      tdm_issue(hi_u, rowBase, k0n, (unsigned)(uintptr_t)&sPan[nxt][0][0]);
      tdm_issue(lo_u, rowBase, k0n, (unsigned)(uintptr_t)&sPan[nxt][1][0]);
      tdm_issue(hi_u, colBase, k0n, (unsigned)(uintptr_t)&sPan[nxt][2][0]);
      tdm_issue(lo_u, colBase, k0n, (unsigned)(uintptr_t)&sPan[nxt][3][0]);
    }

    // A fragment (16x32 bf16): half 0 -> K {0..7,16..23}; half 1 -> K {8..15,24..31}
    Frag ahi[2], alo[2], bhi[4], blo[4];
#pragma unroll
    for (int tm = 0; tm < 2; ++tm) {
      const int m = waveM * 32 + tm * 16 + lm;
      const __bf16* pah = &sPan[cur][0][m * LDT];
      const __bf16* pal = &sPan[cur][1][m * LDT];
      ahi[tm].q[0] = *(const uint4*)(pah + 8 * half);
      ahi[tm].q[1] = *(const uint4*)(pah + 16 + 8 * half);
      alo[tm].q[0] = *(const uint4*)(pal + 8 * half);
      alo[tm].q[1] = *(const uint4*)(pal + 16 + 8 * half);
    }
    // B fragment (32x16 bf16): half 0 -> K {0..7,16..23}; half 1 -> K {8..15,24..31} of row n
#pragma unroll
    for (int tn = 0; tn < 4; ++tn) {
      const int n = waveN * 64 + tn * 16 + lm;
      const __bf16* pbh = &sPan[cur][2][n * LDT];
      const __bf16* pbl = &sPan[cur][3][n * LDT];
      bhi[tn].q[0] = *(const uint4*)(pbh + 8 * half);
      bhi[tn].q[1] = *(const uint4*)(pbh + 16 + 8 * half);
      blo[tn].q[0] = *(const uint4*)(pbl + 8 * half);
      blo[tn].q[1] = *(const uint4*)(pbl + 16 + 8 * half);
    }

    // split-precision accumulate: hi*hi + hi*lo + lo*hi
#pragma unroll
    for (int tm = 0; tm < 2; ++tm)
#pragma unroll
      for (int tn = 0; tn < 4; ++tn) {
        acc[tm][tn] = __builtin_amdgcn_wmma_f32_16x16x32_bf16(
            false, ahi[tm].v, false, bhi[tn].v, (short)0, acc[tm][tn], false, false);
        acc[tm][tn] = __builtin_amdgcn_wmma_f32_16x16x32_bf16(
            false, ahi[tm].v, false, blo[tn].v, (short)0, acc[tm][tn], false, false);
        acc[tm][tn] = __builtin_amdgcn_wmma_f32_16x16x32_bf16(
            false, alo[tm].v, false, bhi[tn].v, (short)0, acc[tm][tn], false, false);
      }

    // wave 0 confirms the prefetch landed, then the block flips buffers
    if ((ks + 1) < NSTEP && wid == 0) __builtin_amdgcn_s_wait_tensorcnt(0);
    __syncthreads();
  }

  // epilogue: Poincare dist^2 -> double sigmoid (C=1, T=1, R=2)
  const float Rv = 2.0f, EPSv = 1e-7f;
#pragma unroll
  for (int tm = 0; tm < 2; ++tm) {
    const int iLoc = waveM * 32 + tm * 16;
    float sqi[8];
#pragma unroll
    for (int r = 0; r < 8; ++r) sqi[r] = sSqR[iLoc + r + 8 * half];
#pragma unroll
    for (int tn = 0; tn < 4; ++tn) {
      const int jLoc = waveN * 64 + tn * 16 + lm;
      const float sqj = sSqC[jLoc];
      const size_t j = (size_t)(colBase + jLoc);
#pragma unroll
      for (int r = 0; r < 8; ++r) {
        const float g  = acc[tm][tn][r];
        const float si = sqi[r];
        float d2  = fmaxf(si + sqj - 2.0f * g, 0.0f);
        float den = fmaxf((1.0f - si) * (1.0f - sqj), EPSv);
        float arg = fmaxf(1.0f + 2.0f * d2 * __builtin_amdgcn_rcpf(den), 1.0f + EPSv);
        float dist  = __logf(arg + __fsqrt_rn(arg * arg - 1.0f));  // arccosh
        float dist2 = dist * dist;
        float p = __builtin_amdgcn_rcpf(1.0f + __expf(dist2 - Rv)); // sigmoid((R-d2)/T)
        float o = __builtin_amdgcn_rcpf(1.0f + __expf(-p));         // sigmoid again
        const size_t i = (size_t)(rowBase + iLoc + r + 8 * half);
        out[i * (size_t)NPTS + j] = o;
      }
    }
  }
}

extern "C" void kernel_launch(void* const* d_in, const int* in_sizes, int n_in,
                              void* d_out, int out_size, void* d_ws, size_t ws_size,
                              hipStream_t stream) {
  (void)in_sizes; (void)n_in; (void)out_size; (void)ws_size;
  const float* x = (const float*)d_in[0];
  float* out = (float*)d_out;

  // workspace layout: hi bf16 (4MB) | lo bf16 (4MB) | sq f32 (32KB)
  unsigned short* hi = (unsigned short*)d_ws;
  unsigned short* lo = hi + (size_t)NPTS * DIM;
  float* sq = (float*)(lo + (size_t)NPTS * DIM);

  prep_kernel<<<NPTS / 8, 256, 0, stream>>>(x, hi, lo, sq);
  fermi_kernel<<<dim3(NPTS / BT, NPTS / BT), 256, 0, stream>>>(hi, lo, sq, out);
}
